// graph_model_42219528520392
// MI455X (gfx1250) — compile-verified, matrix operand order checked by reading, not run on hardware
//
#include <hip/hip_runtime.h>

// ---------------- constants ----------------
#define Bsz   4096
#define Dd    512
#define TASKS 4
#define CLS   65
#define TC    69            // TASKS + CLS
#define NSRC  (TC + Bsz)    // 4165 valid src rows
#define Np    4224          // NSRC padded to multiple of 64 (N-block) and 32 (K-step)
#define CFP   96            // cross_feat columns padded (69 -> 96, multiple of 32)
#define CLSP  128           // class dim padded to multiple of 64
#define TCP   80            // TC padded to multiple of 16

typedef __attribute__((ext_vector_type(16))) __bf16 v16bf;
typedef __attribute__((ext_vector_type(8)))  float  v8f;

union Frag { v16bf v; unsigned int u[8]; };

__device__ __forceinline__ unsigned short f2bf(float f) {
  unsigned int u = __float_as_uint(f);
  u = (u + 0x7FFFu + ((u >> 16) & 1u)) >> 16;   // round-to-nearest-even
  return (unsigned short)u;
}

__device__ __forceinline__ float wave_sum(float v) {
  #pragma unroll
  for (int off = 16; off; off >>= 1) v += __shfl_xor(v, off, 32);
  return v;
}
__device__ __forceinline__ float wave_max(float v) {
  #pragma unroll
  for (int off = 16; off; off >>= 1) v = fmaxf(v, __shfl_xor(v, off, 32));
  return v;
}

// ---------------- spatial mean: x(B,D,7,7) -> xm rows of h ----------------
// Each block reduces 64 consecutive (b,d) rows of 49 elements, staged via LDS
// so global loads are fully coalesced (3136 contiguous floats per block).
// Prefetch the next block's region (emits global_prefetch_b8) to keep the
// 23.3 TB/s HBM stream ahead of the LDS reduction.
__global__ __launch_bounds__(256) void xm_kernel(const float* __restrict__ x,
                                                 float* __restrict__ xm,
                                                 int nblocks) {
  __shared__ float lds[64 * 49];
  long base = (long)blockIdx.x * (64 * 49);
  bool has_next = ((int)blockIdx.x + 1) < nblocks;
  for (int i = threadIdx.x; i < 64 * 49; i += 256) {
    lds[i] = x[base + i];
    if (has_next) __builtin_prefetch(&x[base + 64 * 49 + i], 0, 0);
  }
  __syncthreads();
  if (threadIdx.x < 64) {
    float s = 0.f;
    int o = threadIdx.x * 49;
    #pragma unroll
    for (int j = 0; j < 49; j++) s += lds[o + j];
    xm[(long)blockIdx.x * 64 + threadIdx.x] = s * (1.0f / 49.0f);
  }
}

// ---------------- deterministic segment mean + EMA ----------------
__global__ __launch_bounds__(64) void seg_mean_k(const float* __restrict__ feats,
                                                 const int* __restrict__ ids,
                                                 const float* __restrict__ proto,
                                                 float* __restrict__ out,
                                                 int nseg, int nrows) {
  int seg = blockIdx.x;
  int d = blockIdx.y * 64 + threadIdx.x;
  float s = 0.f, cnt = 0.f;
  for (int b = 0; b < nrows; b++) {
    if (ids[b] == seg) { s += feats[(long)b * Dd + d]; cnt += 1.f; }
  }
  out[seg * Dd + d] = 0.9f * proto[seg * Dd + d] + 0.1f * (s / fmaxf(cnt, 1.f));
}

// ---------------- build src rows into h (f32 + bf16) ----------------
__global__ __launch_bounds__(256) void build_src_k(const float* __restrict__ t1,
                                                   const float* __restrict__ c1,
                                                   float* __restrict__ h,
                                                   unsigned short* __restrict__ hb) {
  int idx = blockIdx.x * 256 + threadIdx.x;   // Np*Dd total (exact)
  int row = idx >> 9, d = idx & 511;
  float v;
  if (row < TASKS)      v = t1[row * Dd + d];
  else if (row < TC)    v = c1[(row - TASKS) * Dd + d];
  else if (row < NSRC)  v = h[idx];           // xm already written here
  else                  v = 0.f;
  h[idx] = v;
  hb[idx] = f2bf(v);
}

// ---------------- transpose-convert f32 -> bf16 ----------------
// out[n*K + k] = bf16(in[k*ldi + n]) for n<Nreal else 0. out dims (Npad x K).
__global__ __launch_bounds__(256) void convT_k(const float* __restrict__ in,
                                               unsigned short* __restrict__ out,
                                               int K, int Nreal, int Npad, int ldi) {
  long idx = (long)blockIdx.x * 256 + threadIdx.x;
  long total = (long)Npad * K;
  if (idx >= total) return;
  int n = (int)(idx / K), k = (int)(idx % K);
  float v = (n < Nreal) ? in[(long)k * ldi + n] : 0.f;
  out[idx] = f2bf(v);
}

// ---------------- plain convert f32 -> bf16 ----------------
__global__ __launch_bounds__(256) void conv_k(const float* __restrict__ in,
                                              unsigned short* __restrict__ out, int n) {
  int i = blockIdx.x * 256 + threadIdx.x;
  if (i < n) out[i] = f2bf(in[i]);
}

__global__ __launch_bounds__(256) void fill_u16_k(unsigned short* p, int n) {
  int i = blockIdx.x * 256 + threadIdx.x;
  if (i < n) p[i] = 0;
}

// ---------------- bf16 WMMA GEMM: C[M,N](f32) = scale * A[M,K] @ Bt[N,K]^T ----------------
// One wave computes a 16 x (16*NBLK) strip: A fragment loaded once per k-step and
// reused across NBLK WMMAs (4x fewer A bytes, back-to-back v_wmma for clausing).
// Fragment K-map (16-bit 16x32): lane idx = lane&15, half = lane>>4;
//   dword v<4 -> K = 2v+8*half ; v>=4 -> K = 16+2(v-4)+8*half.
// C: VGPR i -> row m0 + i + 8*half, col n0 + (lane&15).
template <int NBLK>
__global__ __launch_bounds__(128) void gemm_nt(const unsigned short* __restrict__ A,
                                               const unsigned short* __restrict__ Bt,
                                               float* __restrict__ C,
                                               int M, int N, int K,
                                               int lda, int ldb, int ldc, float scale) {
  int tilesN = N / (16 * NBLK);
  int tiles  = (M >> 4) * tilesN;
  int wid = blockIdx.x * 4 + threadIdx.y;
  if (wid >= tiles) return;                    // uniform per wave: EXEC stays all-1 for WMMA
  int m0 = (wid / tilesN) << 4;
  int n0 = (wid % tilesN) * (16 * NBLK);
  int lane = threadIdx.x;
  int r = lane & 15, half = lane >> 4;
  const unsigned short* ap = A + (size_t)(m0 + r) * lda;
  const unsigned short* bp[NBLK];
  #pragma unroll
  for (int j = 0; j < NBLK; j++)
    bp[j] = Bt + (size_t)(n0 + 16 * j + r) * ldb;
  v8f c[NBLK];
  #pragma unroll
  for (int j = 0; j < NBLK; j++) c[j] = (v8f){};
  for (int k0 = 0; k0 < K; k0 += 32) {
    Frag a;
    #pragma unroll
    for (int v = 0; v < 8; v++) {
      int kk = k0 + ((v < 4) ? (2 * v + 8 * half) : (16 + 2 * (v - 4) + 8 * half));
      a.u[v] = *(const unsigned int*)(ap + kk);
    }
    Frag b[NBLK];
    #pragma unroll
    for (int j = 0; j < NBLK; j++) {
      #pragma unroll
      for (int v = 0; v < 8; v++) {
        int kk = k0 + ((v < 4) ? (2 * v + 8 * half) : (16 + 2 * (v - 4) + 8 * half));
        b[j].u[v] = *(const unsigned int*)(bp[j] + kk);
      }
    }
    #pragma unroll
    for (int j = 0; j < NBLK; j++)
      c[j] = __builtin_amdgcn_wmma_f32_16x16x32_bf16(false, a.v, false, b[j].v,
                                                     (short)0, c[j], false, false);
  }
  #pragma unroll
  for (int j = 0; j < NBLK; j++) {
    float* cp = C + (size_t)(m0 + half * 8) * ldc + n0 + 16 * j + r;
    #pragma unroll
    for (int i = 0; i < 8; i++) cp[(size_t)i * ldc] = c[j][i] * scale;
  }
}

// ---------------- row softmax over 4165 cols, keep first 69 -> cross_feat ----------------
__global__ __launch_bounds__(128) void softmax_cf_k(const float* __restrict__ S,
                                                    float* __restrict__ cf,
                                                    unsigned short* __restrict__ cfb) {
  int b = blockIdx.x * 4 + threadIdx.y;
  if (b >= Bsz) return;
  int lane = threadIdx.x;
  const float* row = S + (size_t)b * Np;
  float mx = -3.4e38f;
  for (int n = lane; n < NSRC; n += 32) mx = fmaxf(mx, row[n]);
  mx = wave_max(mx);
  float s = 0.f;
  for (int n = lane; n < NSRC; n += 32) s += __expf(row[n] - mx);
  s = wave_sum(s);
  float inv = 1.f / s;
  for (int i = lane; i < CFP; i += 32) {
    float v = (i < TC) ? __expf(row[i] - mx) * inv : 0.f;
    cf[(size_t)b * CFP + i] = v;
    cfb[(size_t)b * CFP + i] = f2bf(v);
  }
}

// ---------------- d2[i,j] = ||t1_i - c1_j||^2 ----------------
__global__ __launch_bounds__(128) void d2_k(const float* __restrict__ t1,
                                            const float* __restrict__ c1,
                                            float* __restrict__ d2) {
  int wid = blockIdx.x * 4 + threadIdx.y;
  if (wid >= TASKS * CLS) return;
  int i = wid / CLS, j = wid % CLS, lane = threadIdx.x;
  float s = 0.f;
  for (int d = lane; d < Dd; d += 32) {
    float df = t1[i * Dd + d] - c1[j * Dd + d];
    s += df * df;
  }
  s = wave_sum(s);
  if (lane == 0) d2[i * CLS + j] = s;
}

// ---------------- adj block (69x69) written into A_top bf16 ----------------
__global__ __launch_bounds__(128) void adj_k(const float* __restrict__ t1,
                                             const float* __restrict__ c1,
                                             const float* __restrict__ tw, const float* tb,
                                             const float* __restrict__ cw, const float* cb,
                                             const float* __restrict__ d2,
                                             unsigned short* __restrict__ Atop) {
  int wid = blockIdx.x * 4 + threadIdx.y;
  if (wid >= TC * TC) return;
  int i = wid / TC, j = wid % TC, lane = threadIdx.x;
  float val = 0.f;
  if (i < TASKS && j < TASKS) {
    if (i != j) {
      float s = 0.f;
      for (int d = lane; d < Dd; d += 32)
        s += fabsf(t1[i * Dd + d] - t1[j * Dd + d]) * tw[d];
      s = wave_sum(s) + tb[0];
      val = 1.f / (1.f + __expf(-s));
    }
  } else if (i >= TASKS && j >= TASKS) {
    if (i != j) {
      int ii = i - TASKS, jj = j - TASKS;
      float s = 0.f;
      for (int d = lane; d < Dd; d += 32)
        s += fabsf(c1[ii * Dd + d] - c1[jj * Dd + d]) * cw[d];
      s = wave_sum(s) + cb[0];
      val = 1.f / (1.f + __expf(-s));
    }
  } else {
    int ti = (i < TASKS) ? i : j;
    int cj = (i < TASKS) ? j - TASKS : i - TASKS;
    float m = -3.4e38f;
    #pragma unroll
    for (int t = 0; t < TASKS; t++) m = fmaxf(m, -d2[t * CLS + cj] * (1.f / 16.f));
    float sum = 0.f;
    #pragma unroll
    for (int t = 0; t < TASKS; t++) sum += __expf(-d2[t * CLS + cj] * (1.f / 16.f) - m);
    val = __expf(-d2[ti * CLS + cj] * (1.f / 16.f) - m) / sum;
  }
  if (lane == 0) Atop[(size_t)i * Np + j] = f2bf(val);
}

// ---------------- A_top[:, 69:69+B] = cross_feat^T ----------------
__global__ __launch_bounds__(256) void cfT_k(const float* __restrict__ cf,
                                             unsigned short* __restrict__ Atop) {
  int idx = blockIdx.x * 256 + threadIdx.x;
  if (idx >= TC * Bsz) return;
  int i = idx / Bsz, b = idx % Bsz;
  Atop[(size_t)i * Np + TC + b] = f2bf(cf[(size_t)b * CFP + i]);
}

// ---------------- ae_loss = mean_j sum_i p log p ----------------
__global__ __launch_bounds__(128) void ae_loss_k(const float* __restrict__ d2,
                                                 float* __restrict__ out) {
  __shared__ float sh[128];
  int t = threadIdx.x;
  float s = 0.f;
  if (t < CLS) {
    float e[TASKS];
    float m = -3.4e38f;
    #pragma unroll
    for (int i = 0; i < TASKS; i++) {
      e[i] = -d2[i * CLS + t] * (1.f / 16.f);
      m = fmaxf(m, e[i]);
    }
    float sum = 0.f;
    #pragma unroll
    for (int i = 0; i < TASKS; i++) sum += __expf(e[i] - m);
    #pragma unroll
    for (int i = 0; i < TASKS; i++) {
      float p = __expf(e[i] - m) / sum;
      s += p * __logf(p);
    }
  }
  sh[t] = s;
  __syncthreads();
  for (int off = 64; off; off >>= 1) {
    if (t < off) sh[t] += sh[t + off];
    __syncthreads();
  }
  if (t == 0) out[0] = sh[0] * (1.f / (float)CLS);
}

// ---------------- GNN epilogue: h = relu(m + b), pad rows -> 0 ----------------
__global__ __launch_bounds__(256) void gnn_ep_k(const float* __restrict__ mtop,
                                                const float* __restrict__ mbot,
                                                const float* __restrict__ gb,
                                                float* __restrict__ h,
                                                unsigned short* __restrict__ hb) {
  int idx = blockIdx.x * 256 + threadIdx.x;   // Np*Dd (exact)
  int row = idx >> 9, d = idx & 511;
  float o = 0.f;
  if (row < TC)         o = fmaxf(mtop[row * Dd + d] + gb[d], 0.f);
  else if (row < NSRC)  o = fmaxf(mbot[(size_t)(row - TC) * Dd + d] + gb[d], 0.f);
  h[idx] = o;
  hb[idx] = f2bf(o);
}

// ---------------- final: y = task_cls_w[task] * (x2@W + b); copy labels ----------------
__global__ __launch_bounds__(256) void final_k(const float* __restrict__ ypad,
                                               const float* __restrict__ bb,
                                               const float* __restrict__ tclw,
                                               const int* __restrict__ torder,
                                               const int* __restrict__ labels,
                                               float* __restrict__ out) {
  int idx = blockIdx.x * 256 + threadIdx.x;
  if (idx >= Bsz * CLS) return;
  int b = idx / CLS, c = idx % CLS;
  out[1 + idx] = tclw[torder[b] * CLS + c] * (ypad[(size_t)b * CLSP + c] + bb[c]);
  if (c == 0) ((int*)out)[1 + Bsz * CLS + b] = labels[b];
}

// ---------------- host ----------------
static inline void launch_gemm(const unsigned short* A, const unsigned short* Bt, float* C,
                               int M, int N, int K, int lda, int ldb, int ldc,
                               float scale, hipStream_t s) {
  // all GEMM N dims are multiples of 64 -> 4-wide N blocking
  int tiles = (M / 16) * (N / 64);
  int blocks = (tiles + 3) / 4;
  gemm_nt<4><<<blocks, dim3(32, 4), 0, s>>>(A, Bt, C, M, N, K, lda, ldb, ldc, scale);
}

extern "C" void kernel_launch(void* const* d_in, const int* in_sizes, int n_in,
                              void* d_out, int out_size, void* d_ws, size_t ws_size,
                              hipStream_t stream) {
  const float* x           = (const float*)d_in[0];
  const int*   labels      = (const int*)d_in[1];
  const int*   torder      = (const int*)d_in[2];
  const float* task_rep    = (const float*)d_in[3];
  const float* class_proto = (const float*)d_in[4];
  const float* task_w      = (const float*)d_in[5];
  const float* task_b      = (const float*)d_in[6];
  const float* class_w     = (const float*)d_in[7];
  const float* class_b     = (const float*)d_in[8];
  const float* wq          = (const float*)d_in[9];
  const float* wk          = (const float*)d_in[10];
  const float* gw          = (const float*)d_in[11];
  const float* gb          = (const float*)d_in[12];
  const float* bw          = (const float*)d_in[13];
  const float* bb          = (const float*)d_in[14];
  const float* tclw        = (const float*)d_in[15];
  float* out = (float*)d_out;

  // workspace carve-up
  char* w = (char*)d_ws;
  size_t off = 0;
  auto take = [&](size_t bytes) -> char* {
    off = (off + 255) & ~(size_t)255;
    char* p = w + off;
    off += bytes;
    return p;
  };
  float*          h_f32  = (float*)take((size_t)Np * Dd * 4);
  unsigned short* h_bf16 = (unsigned short*)take((size_t)Np * Dd * 2);
  float*          t1     = (float*)take(TASKS * Dd * 4);
  float*          c1     = (float*)take(CLS * Dd * 4);
  unsigned short* wqT    = (unsigned short*)take((size_t)Dd * Dd * 2);
  unsigned short* wkT    = (unsigned short*)take((size_t)Dd * Dd * 2);
  unsigned short* gwT    = (unsigned short*)take((size_t)2 * Dd * Dd * 2);
  unsigned short* bwT    = (unsigned short*)take((size_t)CLSP * Dd * 2);
  float*          q_f32  = (float*)take((size_t)Bsz * Dd * 4);
  unsigned short* q_bf16 = (unsigned short*)take((size_t)Bsz * Dd * 2);
  float*          k_f32  = (float*)take((size_t)Np * Dd * 4);
  unsigned short* k_bf16 = (unsigned short*)take((size_t)Np * Dd * 2);
  float*          scores = (float*)take((size_t)Bsz * Np * 4);
  float*          cf     = (float*)take((size_t)Bsz * CFP * 4);
  unsigned short* cfb    = (unsigned short*)take((size_t)Bsz * CFP * 2);
  float*          d2buf  = (float*)take(TASKS * CLS * 4);
  unsigned short* Atop   = (unsigned short*)take((size_t)TCP * Np * 2);
  float*          hw     = (float*)take((size_t)Np * Dd * 4);
  unsigned short* hwT    = (unsigned short*)take((size_t)Dd * Np * 2);
  float*          mtop   = (float*)take((size_t)TCP * Dd * 4);
  float*          mbot   = (float*)take((size_t)Bsz * Dd * 4);
  float*          ypad   = (float*)take((size_t)Bsz * CLSP * 4);

  float* xm = h_f32 + (size_t)TC * Dd;   // xm lives in h rows [69, 69+B)

  // 1) spatial mean (memory-bound stage: 411 MB of x)
  int xblocks = (Bsz * Dd) / 64;
  xm_kernel<<<xblocks, 256, 0, stream>>>(x, xm, xblocks);

  // 2) deterministic segment means -> t1, c1
  seg_mean_k<<<dim3(TASKS, Dd / 64), 64, 0, stream>>>(xm, torder, task_rep, t1, TASKS, Bsz);
  seg_mean_k<<<dim3(CLS, Dd / 64), 64, 0, stream>>>(xm, labels, class_proto, c1, CLS, Bsz);

  // 3) assemble src into h (f32 + bf16), zero pad rows
  build_src_k<<<(Np * Dd) / 256, 256, 0, stream>>>(t1, c1, h_f32, h_bf16);

  // 4) weights -> transposed bf16 (Bt layout for the WMMA GEMM)
  convT_k<<<((Dd * Dd) + 255) / 256, 256, 0, stream>>>(wq, wqT, Dd, Dd, Dd, Dd);
  convT_k<<<((Dd * Dd) + 255) / 256, 256, 0, stream>>>(wk, wkT, Dd, Dd, Dd, Dd);
  convT_k<<<((Dd * Dd) + 255) / 256, 256, 0, stream>>>(gw, gwT, Dd, Dd, Dd, Dd);
  convT_k<<<((Dd * Dd) + 255) / 256, 256, 0, stream>>>(gw + Dd * Dd, gwT + Dd * Dd, Dd, Dd, Dd, Dd);
  convT_k<<<((CLSP * Dd) + 255) / 256, 256, 0, stream>>>(bw, bwT, Dd, CLS, CLSP, CLS);

  // 5) q = xm @ Wq ; k = src @ Wk  (bf16 WMMA, f32 accum)
  launch_gemm(h_bf16 + (size_t)TC * Dd, wqT, q_f32, Bsz, Dd, Dd, Dd, Dd, Dd, 1.0f, stream);
  launch_gemm(h_bf16, wkT, k_f32, Np, Dd, Dd, Dd, Dd, Dd, 1.0f, stream);
  conv_k<<<((Bsz * Dd) + 255) / 256, 256, 0, stream>>>(q_f32, q_bf16, Bsz * Dd);
  conv_k<<<((Np * Dd) + 255) / 256, 256, 0, stream>>>(k_f32, k_bf16, Np * Dd);

  // 6) scores = (q @ k^T) / sqrt(D)   (k row-major IS Bt layout for q@k^T)
  launch_gemm(q_bf16, k_bf16, scores, Bsz, Np, Dd, Dd, Dd, Np,
              0.04419417382415922f, stream);

  // 7) row softmax over 4165 valid cols -> cross_feat (first 69 cols)
  softmax_cf_k<<<Bsz / 4, dim3(32, 4), 0, stream>>>(scores, cf, cfb);

  // 8) graph blocks: d2, adj (tg/cg/cross_graph) into A_top, cross_feat^T into A_top
  d2_k<<<(TASKS * CLS + 3) / 4, dim3(32, 4), 0, stream>>>(t1, c1, d2buf);
  fill_u16_k<<<((TCP * Np) + 255) / 256, 256, 0, stream>>>(Atop, TCP * Np);
  adj_k<<<(TC * TC + 3) / 4, dim3(32, 4), 0, stream>>>(t1, c1, task_w, task_b,
                                                       class_w, class_b, d2buf, Atop);
  cfT_k<<<((TC * Bsz) + 255) / 256, 256, 0, stream>>>(cf, Atop);

  // 9) ae_loss -> out[0]
  ae_loss_k<<<1, 128, 0, stream>>>(d2buf, out);

  // 10) GNN layers, exploiting adj_all block sparsity:
  //     top rows:  A_top(80 x 4224) @ hw          (needs hw^T bf16)
  //     bot rows:  cf(4096 x 96)    @ hw[0:69]    (zero-padded cols kill garbage)
  for (int l = 0; l < 2; l++) {
    launch_gemm(h_bf16, gwT + (size_t)l * Dd * Dd, hw, Np, Dd, Dd, Dd, Dd, Dd, 1.0f, stream);
    convT_k<<<(((size_t)Dd * Np) + 255) / 256, 256, 0, stream>>>(hw, hwT, Np, Dd, Dd, Dd);
    launch_gemm(Atop, hwT, mtop, TCP, Dd, Np, Np, Np, Dd, 1.0f, stream);
    launch_gemm(cfb, hwT, mbot, Bsz, Dd, CFP, CFP, Np, Dd, 1.0f, stream);
    gnn_ep_k<<<(Np * Dd) / 256, 256, 0, stream>>>(mtop, mbot, gb + l * Dd, h_f32, h_bf16);
  }

  // 11) y = x2 @ base_w (padded to 128 cols), then scale + bias + label copy
  launch_gemm(h_bf16 + (size_t)TC * Dd, bwT, ypad, Bsz, CLSP, Dd, Dd, Dd, CLSP, 1.0f, stream);
  final_k<<<((Bsz * CLS) + 255) / 256, 256, 0, stream>>>(ypad, bb, tclw, torder, labels, out);
}
